// TransformerChannelAttention_1821066133519
// MI455X (gfx1250) — matrix operand order repaired and checked by reading, not compiled, not run on hardware
//
#include <hip/hip_runtime.h>
#include <hip/hip_bf16.h>
#include <stdint.h>

// ---------- types for CDNA5 WMMA ----------
typedef __bf16 bf16;
typedef __attribute__((ext_vector_type(16))) __bf16 v16bf;
typedef __attribute__((ext_vector_type(8)))  float  v8f;
typedef unsigned long long u64t;

#define C_DIM   128
#define HEADS   8
#define S_DIM   3
#define H_DIM   128
#define W_DIM   128
#define B_DIM   2
#define TTILE   32                       // tokens per tile
#define LDX     132                      // padded bf16 row (bank-conflict-free)
#define LDQ     130                      // padded f32 row
#define NTOK    (H_DIM*W_DIM*S_DIM)      // 49152
#define TILES_PER_B (H_DIM*S_DIM*(W_DIM/TTILE))   // 1536
#define PART_PER_TILE (HEADS*288)        // per-head: G[256] + sqq[16] + sqk[16]

union AFrag { v16bf v; u64t d[4]; };
union BFrag { v16bf v; u64t d[4]; };

// ---------- K0: convert Wq/Wk/Wv to bf16 (row-major, B-matrix friendly) ----------
__global__ void k_cvt_w(const float* __restrict__ wq, const float* __restrict__ wk,
                        const float* __restrict__ wv, bf16* __restrict__ wbf) {
    int i = blockIdx.x * 256 + threadIdx.x;
    if (i < 3 * C_DIM * C_DIM) {
        const float* src = (i < 16384) ? wq : (i < 32768 ? wk : wv);
        wbf[i] = (bf16)src[i & 16383];
    }
}

// ---------- K1: projections (WMMA) + v store + Gram/sumsq partials ----------
__global__ void __launch_bounds__(256) k_proj(const float* __restrict__ x,
                                              const bf16* __restrict__ wbf,
                                              bf16*  __restrict__ vbuf,
                                              float* __restrict__ pbuf) {
    __shared__ bf16  sX[TTILE * LDX];   // staged activations (bf16)
    __shared__ float sQ[TTILE * LDQ];
    __shared__ float sK[TTILE * LDQ];
    __shared__ bf16  sV[TTILE * LDX];

    const int tid = threadIdx.x;
    int bid = blockIdx.x;
    const int wt    = bid & 3;  bid >>= 2;
    const int s_idx = bid % S_DIM; bid /= S_DIM;
    const int h_idx = bid & (H_DIM - 1);
    const int b     = bid >> 7;
    const int w0    = wt * TTILE;

    __builtin_prefetch(wbf, 0, 0);   // global_prefetch_b8: pull weights toward L2/L0

    // ---- stage x[b][c][s][h][w0..w0+31] -> sX[token][channel] (bf16) ----
    const size_t cstride = (size_t)S_DIM * H_DIM * W_DIM;
    const size_t xbase = ((((size_t)b * C_DIM) * S_DIM + s_idx) * H_DIM + h_idx) * W_DIM + w0;
    #pragma unroll
    for (int it = 0; it < 16; ++it) {
        int lin = it * 256 + tid;
        int c  = lin >> 5;
        int wi = lin & 31;
        sX[wi * LDX + c] = (bf16)x[xbase + (size_t)c * cstride + wi];
    }
    __syncthreads();

    const int wv_id = tid >> 5;         // wave32!
    const int lane  = tid & 31;
    const int hi    = lane >> 4;
    const int lo    = lane & 15;
    const int m     = wv_id & 1;        // 2 M-blocks of 16 tokens
    const int ngrp  = wv_id >> 1;       // 4 groups x 2 N-blocks
    const int tokRow = m * 16 + lo;
    const int mrow   = m * 16 + hi * 8; // C/D frag: M = r + 8*(lane>=16)

    // A-frags: ISA 16-bit A layout -> lane holds K = grp*16 + hi*8 + {0..7}
    AFrag a[4];
    #pragma unroll
    for (int kc = 0; kc < 4; ++kc) {
        int b0 = tokRow * LDX + kc * 32 + hi * 8;
        a[kc].d[0] = *(const u64t*)&sX[b0];
        a[kc].d[1] = *(const u64t*)&sX[b0 + 4];
        a[kc].d[2] = *(const u64t*)&sX[b0 + 16];
        a[kc].d[3] = *(const u64t*)&sX[b0 + 20];
    }

    #pragma unroll
    for (int proj = 0; proj < 3; ++proj) {
        const bf16* wp = wbf + proj * (C_DIM * C_DIM);
        #pragma unroll
        for (int nn = 0; nn < 2; ++nn) {
            const int nblk = ngrp * 2 + nn;
            v8f c = {0.f,0.f,0.f,0.f,0.f,0.f,0.f,0.f};
            #pragma unroll
            for (int kc = 0; kc < 4; ++kc) {
                // B[k=i][n=o] = W[o][i]; lane: col o = lo, rows kc*32 + hi*8 + {0..7, 16..23}
                BFrag bf_;
                const bf16* bp = wp + (nblk * 16 + lo) * C_DIM + kc * 32 + hi * 8;
                bf_.d[0] = *(const u64t*)(bp);
                bf_.d[1] = *(const u64t*)(bp + 4);
                bf_.d[2] = *(const u64t*)(bp + 16);
                bf_.d[3] = *(const u64t*)(bp + 20);
                c = __builtin_amdgcn_wmma_f32_16x16x32_bf16(
                        false, a[kc].v, false, bf_.v, (short)0, c, false, false);
            }
            const int ch = nblk * 16 + lo;
            if (proj == 0) {
                #pragma unroll
                for (int r = 0; r < 8; ++r) sQ[(mrow + r) * LDQ + ch] = c[r];
            } else if (proj == 1) {
                #pragma unroll
                for (int r = 0; r < 8; ++r) sK[(mrow + r) * LDQ + ch] = c[r];
            } else {
                #pragma unroll
                for (int r = 0; r < 8; ++r) sV[(mrow + r) * LDX + ch] = (bf16)c[r];
            }
        }
    }
    __syncthreads();

    const int tileIdx = (h_idx * S_DIM + s_idx) * (W_DIM / TTILE) + wt;
    float* pout = pbuf + ((size_t)(b * TILES_PER_B + tileIdx)) * PART_PER_TILE;

    // ---- raw Gram partial: G[h][d][e] = sum_t K[t,hd+d]*Q[t,hd+e] ----
    {
        const int h   = tid >> 5;
        const int sub = tid & 31;
        float acc[8] = {0.f,0.f,0.f,0.f,0.f,0.f,0.f,0.f};
        for (int t = 0; t < TTILE; ++t) {
            const float* kRow = &sK[t * LDQ + h * 16];
            const float* qRow = &sQ[t * LDQ + h * 16];
            #pragma unroll
            for (int p = 0; p < 8; ++p) {
                int lin2 = sub * 8 + p;
                acc[p] += kRow[lin2 >> 4] * qRow[lin2 & 15];
            }
        }
        #pragma unroll
        for (int p = 0; p < 8; ++p) pout[h * 288 + sub * 8 + p] = acc[p];
    }
    // ---- sum-of-squares partials for the L2 norms ----
    {
        const int  ch  = tid & 127;
        const bool isq = tid < 128;
        float s = 0.f;
        for (int t = 0; t < TTILE; ++t) {
            float u = isq ? sQ[t * LDQ + ch] : sK[t * LDQ + ch];
            s += u * u;
        }
        pout[(ch >> 4) * 288 + (isq ? 256 : 272) + (ch & 15)] = s;
    }
    // ---- coalesced v store (bf16) ----
    {
        const size_t ubase = ((size_t)(h_idx * S_DIM + s_idx)) * W_DIM + w0;
        unsigned int* vbu = (unsigned int*)vbuf;
        const unsigned int* sVu = (const unsigned int*)sV;
        const size_t vrowbase = ((size_t)b * NTOK + ubase) * 64;
        #pragma unroll
        for (int it = 0; it < 8; ++it) {
            int lin = it * 256 + tid;
            int tok = lin >> 6, cp = lin & 63;
            vbu[vrowbase + (size_t)tok * 64 + cp] = sVu[tok * (LDX / 2) + cp];
        }
    }
}

// ---------- K2: reduce partials, normalize, softmax, fold into M = Wp*BD(attn) ----------
__global__ void __launch_bounds__(256) k_attn(const float* __restrict__ pbuf,
                                              const float* __restrict__ wpmat,
                                              const float* __restrict__ rescale,
                                              bf16* __restrict__ mbf) {
    __shared__ float acc[PART_PER_TILE];
    __shared__ float attn[HEADS * 256];
    const int b = blockIdx.x;
    const int tid = threadIdx.x;
    const float* pb = pbuf + (size_t)b * TILES_PER_B * PART_PER_TILE;

    for (int idx = tid; idx < PART_PER_TILE; idx += 256) {
        float s = 0.f;
        for (int t = 0; t < TILES_PER_B; ++t) s += pb[(size_t)t * PART_PER_TILE + idx];
        acc[idx] = s;
    }
    __syncthreads();

    if (tid < 128) {                    // one thread per (head, d) row
        const int h = tid >> 4, d = tid & 15;
        const float nk = fmaxf(sqrtf(acc[h * 288 + 272 + d]), 1e-12f);
        const float rs = rescale[h];
        float row[16], mx = -3.4e38f;
        #pragma unroll
        for (int e = 0; e < 16; ++e) {
            float nq = fmaxf(sqrtf(acc[h * 288 + 256 + e]), 1e-12f);
            float v = acc[h * 288 + d * 16 + e] / (nk * nq) * rs;
            row[e] = v; mx = fmaxf(mx, v);
        }
        float sum = 0.f;
        #pragma unroll
        for (int e = 0; e < 16; ++e) { row[e] = __expf(row[e] - mx); sum += row[e]; }
        const float inv = 1.f / sum;
        #pragma unroll
        for (int e = 0; e < 16; ++e) attn[h * 256 + d * 16 + e] = row[e] * inv;
    }
    __syncthreads();

    // M[i][j] = sum_d Wp[i, h*16+d] * attn_h[d, e],  j = h*16+e
    for (int r = 0; r < 64; ++r) {
        int lin = r * 256 + tid;
        int i = lin >> 7, j = lin & 127;
        int h = j >> 4,  e = j & 15;
        float s = 0.f;
        #pragma unroll
        for (int d = 0; d < 16; ++d)
            s += wpmat[i * C_DIM + h * 16 + d] * attn[h * 256 + d * 16 + e];
        mbf[(size_t)b * C_DIM * C_DIM + lin] = (bf16)s;
    }
}

// ---------- K3: final = v @ M^T + bp (WMMA), fused with output transpose ----------
__global__ void __launch_bounds__(256) k_out(const bf16* __restrict__ vbuf,
                                             const bf16* __restrict__ mbf,
                                             const float* __restrict__ bias,
                                             float* __restrict__ out) {
    __shared__ bf16  sVt[TTILE * LDX];
    __shared__ float sO [TTILE * LDQ];
    __shared__ float sB [C_DIM];

    const int tid = threadIdx.x;
    int bid = blockIdx.x;
    const int wt    = bid & 3;  bid >>= 2;
    const int s_idx = bid % S_DIM; bid /= S_DIM;
    const int h_idx = bid & (H_DIM - 1);
    const int b     = bid >> 7;
    const int w0    = wt * TTILE;

    __builtin_prefetch(mbf + (size_t)b * C_DIM * C_DIM, 0, 0);

    // stage v tile (coalesced dword copies)
    {
        const size_t ubase = ((size_t)(h_idx * S_DIM + s_idx)) * W_DIM + w0;
        const unsigned int* vbu = (const unsigned int*)vbuf;
        unsigned int* sVu = (unsigned int*)sVt;
        const size_t vrowbase = ((size_t)b * NTOK + ubase) * 64;
        #pragma unroll
        for (int it = 0; it < 8; ++it) {
            int lin = it * 256 + tid;
            int tok = lin >> 6, cp = lin & 63;
            sVu[tok * (LDX / 2) + cp] = vbu[vrowbase + (size_t)tok * 64 + cp];
        }
    }
    if (tid < 128) sB[tid] = bias[tid];
    __syncthreads();

    const int wv_id = tid >> 5, lane = tid & 31, hi = lane >> 4, lo = lane & 15;
    const int m = wv_id & 1, ngrp = wv_id >> 1;
    const int tokRow = m * 16 + lo;
    const int mrow   = m * 16 + hi * 8;

    AFrag a[4];
    #pragma unroll
    for (int kc = 0; kc < 4; ++kc) {
        int b0 = tokRow * LDX + kc * 32 + hi * 8;
        a[kc].d[0] = *(const u64t*)&sVt[b0];
        a[kc].d[1] = *(const u64t*)&sVt[b0 + 4];
        a[kc].d[2] = *(const u64t*)&sVt[b0 + 16];
        a[kc].d[3] = *(const u64t*)&sVt[b0 + 20];
    }

    const bf16* mb = mbf + (size_t)b * C_DIM * C_DIM;
    #pragma unroll
    for (int nn = 0; nn < 2; ++nn) {
        const int nblk = ngrp * 2 + nn;
        v8f c = {0.f,0.f,0.f,0.f,0.f,0.f,0.f,0.f};
        #pragma unroll
        for (int kc = 0; kc < 4; ++kc) {
            BFrag bf_;
            const bf16* bp = mb + (nblk * 16 + lo) * C_DIM + kc * 32 + hi * 8;
            bf_.d[0] = *(const u64t*)(bp);
            bf_.d[1] = *(const u64t*)(bp + 4);
            bf_.d[2] = *(const u64t*)(bp + 16);
            bf_.d[3] = *(const u64t*)(bp + 20);
            c = __builtin_amdgcn_wmma_f32_16x16x32_bf16(
                    false, a[kc].v, false, bf_.v, (short)0, c, false, false);
        }
        const int ch = nblk * 16 + lo;
        const float bv = sB[ch];
        #pragma unroll
        for (int r = 0; r < 8; ++r) sO[(mrow + r) * LDQ + ch] = c[r] + bv;
    }
    __syncthreads();

    // out[b][i][s_idx][h_idx][w0+tok]  — contiguous in tok per channel i
    #pragma unroll
    for (int it = 0; it < 16; ++it) {
        int lin = it * 256 + tid;
        int i = lin >> 5, tok = lin & 31;
        size_t addr = ((((size_t)b * C_DIM + i) * S_DIM + s_idx) * H_DIM + h_idx) * W_DIM + w0 + tok;
        out[addr] = sO[tok * LDQ + i];
    }
}

extern "C" void kernel_launch(void* const* d_in, const int* in_sizes, int n_in,
                              void* d_out, int out_size, void* d_ws, size_t ws_size,
                              hipStream_t stream) {
    const float* x       = (const float*)d_in[0];
    const float* wq      = (const float*)d_in[1];
    const float* wk      = (const float*)d_in[2];
    const float* wv      = (const float*)d_in[3];
    const float* wp      = (const float*)d_in[4];
    const float* bp      = (const float*)d_in[5];
    const float* rescale = (const float*)d_in[6];
    float* out = (float*)d_out;

    // workspace layout (bytes)
    char* ws = (char*)d_ws;
    const size_t vbytes = (size_t)B_DIM * NTOK * C_DIM * sizeof(bf16);           // 25,165,824
    const size_t pbytes = (size_t)B_DIM * TILES_PER_B * PART_PER_TILE * 4;       // 28,311,552
    const size_t mbytes = (size_t)B_DIM * C_DIM * C_DIM * sizeof(bf16);          //     65,536
    bf16*  vbuf = (bf16*)ws;
    float* pbuf = (float*)(ws + vbytes);
    bf16*  mbf  = (bf16*)(ws + vbytes + pbytes);
    bf16*  wbf  = (bf16*)(ws + vbytes + pbytes + mbytes);

    k_cvt_w<<<(3 * C_DIM * C_DIM + 255) / 256, 256, 0, stream>>>(wq, wk, wv, wbf);
    k_proj<<<B_DIM * TILES_PER_B, 256, 0, stream>>>(x, wbf, vbuf, pbuf);
    k_attn<<<B_DIM, 256, 0, stream>>>(pbuf, wp, rescale, mbf);
    k_out <<<B_DIM * TILES_PER_B, 256, 0, stream>>>(vbuf, mbf, bp, out);
}
